// SampleScoreModel_45019847197458
// MI455X (gfx1250) — matrix operand order repaired and checked by reading, not compiled, not run on hardware
//
#include <hip/hip_runtime.h>
#include <math.h>

typedef __attribute__((ext_vector_type(16))) __bf16 v16bf;
typedef __attribute__((ext_vector_type(8)))  float  v8f;
typedef int v4i_vs __attribute__((vector_size(16)));   // matches builtin param type

#define BM   32      // query rows per workgroup
#define BN   128     // samples per tile
#define DD   256     // feature dim
#define KPAD 264     // row stride (bf16) row-major K tile  (528B, 16B aligned)
#define VPAD 136     // row stride (bf16) transposed V tile (272B, 16B aligned)
#define SSPAD 128    // row stride (f32) S tile
#define PPAD 136     // row stride (bf16) P tile            (272B, 16B aligned)
#define LN_RATIO 9.210340371976184f   // ln(100.0/0.01)

// ---------------- CDNA5 async global->LDS copy ----------------
#if defined(__HIP_DEVICE_COMPILE__) && __has_builtin(__builtin_amdgcn_global_load_async_to_lds_b128)
#define HAVE_ASYNC_LDS 1
#endif

__device__ __forceinline__ void async_copy16(const void* g, void* l) {
#ifdef HAVE_ASYNC_LDS
    __builtin_amdgcn_global_load_async_to_lds_b128(
        (__attribute__((address_space(1))) v4i_vs*)(uintptr_t)g,
        (__attribute__((address_space(3))) v4i_vs*)(uint32_t)(uintptr_t)l,
        0, 0);
#else
    *(uint4*)l = *(const uint4*)g;
#endif
}
__device__ __forceinline__ void async_wait_all() {
#ifdef HAVE_ASYNC_LDS
    asm volatile("s_wait_asynccnt 0" ::: "memory");
#endif
}
// Wait until at most 32 async ops outstanding (the 32 just-issued prefetches
// for the next tile); in-order retirement means the current tile is complete.
__device__ __forceinline__ void async_wait_prev() {
#ifdef HAVE_ASYNC_LDS
    asm volatile("s_wait_asynccnt 32" ::: "memory");
#endif
}

// ---------------- pre-pass: samples -> bf16 (row-major + transposed) + ||s||^2 ----------------
__global__ __launch_bounds__(256, 1)
void prepass_kernel(const float* __restrict__ samples,
                    __bf16* __restrict__ Kg,      // [N][256] bf16
                    __bf16* __restrict__ KTg,     // [256][N] bf16
                    float*  __restrict__ s2g,     // [N]
                    int Nv)
{
    __shared__ __bf16 Tt[64][72];   // [c_local][n_local] transpose tile
    __shared__ float  s2l[64];

    const int tid = threadIdx.x;
    const int n0  = blockIdx.x * 64;

    if (tid < 64) s2l[tid] = 0.0f;
    __syncthreads();

    for (int ct = 0; ct < 4; ++ct) {
        const int c0 = ct * 64;
#pragma unroll
        for (int k = 0; k < 4; ++k) {
            const int i  = tid + k * 256;
            const int r  = i >> 4;          // n local 0..63
            const int cq = (i & 15) * 4;    // c local 0..60
            const float4 f4 = *(const float4*)(samples + (size_t)(n0 + r) * DD + c0 + cq);
            const __bf16 b0 = (__bf16)f4.x, b1 = (__bf16)f4.y;
            const __bf16 b2 = (__bf16)f4.z, b3 = (__bf16)f4.w;
            __bf16* kp = Kg + (size_t)(n0 + r) * DD + c0 + cq;
            kp[0] = b0; kp[1] = b1; kp[2] = b2; kp[3] = b3;
            Tt[cq + 0][r] = b0; Tt[cq + 1][r] = b1;
            Tt[cq + 2][r] = b2; Tt[cq + 3][r] = b3;
            atomicAdd(&s2l[r], f4.x*f4.x + f4.y*f4.y + f4.z*f4.z + f4.w*f4.w);
        }
        __syncthreads();
#pragma unroll
        for (int k = 0; k < 4; ++k) {
            const int i  = tid + k * 256;
            const int cl = i >> 4;          // c local 0..63
            const int nq = (i & 15) * 4;    // n local 0..60
            __bf16* dst = KTg + (size_t)(c0 + cl) * Nv + n0 + nq;
            dst[0] = Tt[cl][nq + 0]; dst[1] = Tt[cl][nq + 1];
            dst[2] = Tt[cl][nq + 2]; dst[3] = Tt[cl][nq + 3];
        }
        __syncthreads();
    }
    if (tid < 64) s2g[n0 + tid] = s2l[tid];
}

// ---------------- main flash-softmax kernel (double-buffered async pipeline) ----------------
__global__ __launch_bounds__(256, 1)
void score_kernel(const float* __restrict__ tarr,
                  const float* __restrict__ x,
                  const __bf16* __restrict__ Kg,
                  const __bf16* __restrict__ KTg,
                  const float* __restrict__ s2g,
                  float* __restrict__ out, int Nv)
{
    __shared__ __bf16 Ks [2 * BN * KPAD];   // 135,168 B
    __shared__ __bf16 VTs[2 * DD * VPAD];   // 139,264 B
    __shared__ float  Ss [BM * SSPAD];      //  16,384 B
    __shared__ __bf16 Ps [BM * PPAD];       //   8,704 B
    __shared__ float  s2s[2 * BN];          //   1,024 B
    __shared__ float  mrow[BM], lrow[BM], arow[BM], invden[BM], tsc[BM]; // 640 B

    const int tid  = threadIdx.x;
    const int wave = tid >> 5;
    const int lane = tid & 31;
    const int m0   = blockIdx.x * BM;

    const int mb    = wave & 1;
    const int dgrp  = wave >> 1;
    const int hi    = (lane >= 16) ? 1 : 0;
    const int l16   = lane & 15;
    const int rA    = mb * 16 + l16;
    const int koffA = hi * 8;

    // per-tile async fill: 16 + 16 async-b128 instructions per wave
    auto issue_tile = [&](int n0t, int buf) {
        __bf16* KsB = Ks  + buf * (BN * KPAD);
        __bf16* VTB = VTs + buf * (DD * VPAD);
        for (int i = tid; i < BN * DD / 8; i += 256) {
            const int flat = i * 8;
            const int n = flat >> 8;
            const int c = flat & 255;
            async_copy16(Kg + (size_t)(n0t + n) * DD + c, KsB + n * KPAD + c);
        }
        for (int i = tid; i < DD * BN / 8; i += 256) {
            const int flat = i * 8;
            const int d  = flat >> 7;
            const int nn = flat & 127;
            async_copy16(KTg + (size_t)d * Nv + n0t + nn, VTB + d * VPAD + nn);
        }
        if (tid < BN) s2s[buf * BN + tid] = s2g[n0t + tid];
    };

    if (tid < BM) {
        float tv = tarr[m0 + tid];
        float ts = 0.01f * __expf(tv * LN_RATIO);
        tsc[tid]    = ts;
        invden[tid] = 1.0f / (ts * ts + 1e-8f);
        mrow[tid]   = -__builtin_inff();
        lrow[tid]   = 0.0f;
    }

    // prologue: prefetch tile 0 into buffer 0
    issue_tile(0, 0);

    // Q A-fragments in registers (bf16), pre-swizzled for WMMA
    v16bf Qa[8];
    {
        const float* qrow = x + (size_t)(m0 + rA) * DD;
#pragma unroll
        for (int kk = 0; kk < 8; ++kk) {
            const int k0 = kk * 32;
#pragma unroll
            for (int e = 0; e < 8; ++e) {
                Qa[kk][e]     = (__bf16)qrow[k0 + 2 * koffA + e];
                Qa[kk][e + 8] = (__bf16)qrow[k0 + 2 * koffA + 8 + e];
            }
        }
    }

    v8f Oacc[4];
    {
        v8f z = {};
#pragma unroll
        for (int dt = 0; dt < 4; ++dt) Oacc[dt] = z;
    }

    const int ncol0 = dgrp * 32;
    const int T = Nv / BN;

    for (int it = 0; it < T; ++it) {
        const int cur = it & 1;

        // prefetch next tile into the other buffer (safe: that buffer was
        // last read in iteration it-1, sealed by its end-of-loop barrier)
        if (it + 1 < T) {
            issue_tile((it + 1) * BN, cur ^ 1);
            async_wait_prev();    // drain current tile; next tile stays in flight
        } else {
            async_wait_all();
        }
        __syncthreads();

        const __bf16* KsB = Ks  + cur * (BN * KPAD);
        const __bf16* VTB = VTs + cur * (DD * VPAD);
        const float*  s2B = s2s + cur * BN;

        // ---- S = Q * K^T : two 16x16 tiles per wave, K=256 in 8 WMMA steps ----
#pragma unroll
        for (int tt = 0; tt < 2; ++tt) {
            const int ncol = ncol0 + tt * 16;
            v8f sacc = {};
            const __bf16* brow = KsB + (ncol + l16) * KPAD + hi * 16;
#pragma unroll
            for (int kk = 0; kk < 8; ++kk) {
                v16bf b;
                const __bf16* bp = brow + kk * 32;
#pragma unroll
                for (int e = 0; e < 16; ++e) b[e] = bp[e];
                sacc = __builtin_amdgcn_wmma_f32_16x16x32_bf16(
                           false, Qa[kk], false, b, (short)0, sacc, false, false);
            }
#pragma unroll
            for (int vv = 0; vv < 8; ++vv)
                Ss[(mb * 16 + vv + hi * 8) * SSPAD + ncol + l16] = sacc[vv];
        }
        __syncthreads();

        // ---- online softmax: each wave owns 4 complete rows ----
#pragma unroll
        for (int rr = 0; rr < 4; ++rr) {
            const int r = wave * 4 + rr;
            const float iv = invden[r];
            float v0 = (Ss[r * SSPAD + lane     ] - 0.5f * s2B[lane     ]) * iv;
            float v1 = (Ss[r * SSPAD + lane + 32] - 0.5f * s2B[lane + 32]) * iv;
            float v2 = (Ss[r * SSPAD + lane + 64] - 0.5f * s2B[lane + 64]) * iv;
            float v3 = (Ss[r * SSPAD + lane + 96] - 0.5f * s2B[lane + 96]) * iv;
            float tm = fmaxf(fmaxf(v0, v1), fmaxf(v2, v3));
#pragma unroll
            for (int off = 16; off >= 1; off >>= 1)
                tm = fmaxf(tm, __shfl_xor(tm, off, 32));
            const float mold = mrow[r];
            const float mnew = fmaxf(mold, tm);
            const float p0 = __expf(v0 - mnew), p1 = __expf(v1 - mnew);
            const float p2 = __expf(v2 - mnew), p3 = __expf(v3 - mnew);
            float ls = p0 + p1 + p2 + p3;
#pragma unroll
            for (int off = 16; off >= 1; off >>= 1)
                ls += __shfl_xor(ls, off, 32);
            if (lane == 0) {
                const float al = __expf(mold - mnew);
                lrow[r] = lrow[r] * al + ls;
                mrow[r] = mnew;
                arow[r] = al;
            }
            Ps[r * PPAD + lane     ] = (__bf16)p0;
            Ps[r * PPAD + lane + 32] = (__bf16)p1;
            Ps[r * PPAD + lane + 64] = (__bf16)p2;
            Ps[r * PPAD + lane + 96] = (__bf16)p3;
        }
        __syncthreads();

        // ---- rescale O and accumulate O += P * V ----
        float al[8];
#pragma unroll
        for (int vv = 0; vv < 8; ++vv) al[vv] = arow[mb * 16 + vv + hi * 8];
#pragma unroll
        for (int dt = 0; dt < 4; ++dt)
#pragma unroll
            for (int vv = 0; vv < 8; ++vv) Oacc[dt][vv] *= al[vv];

#pragma unroll
        for (int ks = 0; ks < 4; ++ks) {
            v16bf a;
            const __bf16* ap = &Ps[rA * PPAD + ks * 32 + 2 * koffA];
#pragma unroll
            for (int e = 0; e < 8; ++e) { a[e] = ap[e]; a[e + 8] = ap[e + 8]; }
#pragma unroll
            for (int dt = 0; dt < 4; ++dt) {
                const int d = dgrp * 64 + dt * 16 + l16;
                const __bf16* bp = VTB + d * VPAD + ks * 32 + hi * 16;
                v16bf b;
#pragma unroll
                for (int e = 0; e < 16; ++e) b[e] = bp[e];
                Oacc[dt] = __builtin_amdgcn_wmma_f32_16x16x32_bf16(
                               false, a, false, b, (short)0, Oacc[dt], false, false);
            }
        }
        __syncthreads();
    }

    // ---- epilogue: out = t_scale * (O/l - x) / denom ----
#pragma unroll
    for (int dt = 0; dt < 4; ++dt) {
        const int d = dgrp * 64 + dt * 16 + l16;
#pragma unroll
        for (int vv = 0; vv < 8; ++vv) {
            const int r = mb * 16 + vv + hi * 8;
            const float mean = Oacc[dt][vv] / lrow[r];
            const float xv = x[(size_t)(m0 + r) * DD + d];
            out[(size_t)(m0 + r) * DD + d] = tsc[r] * (mean - xv) * invden[r];
        }
    }
}

// ---------------- fallback (self-contained, no workspace) ----------------
__global__ __launch_bounds__(256, 1)
void score_kernel_fallback(const float* __restrict__ tarr,
                           const float* __restrict__ x,
                           const float* __restrict__ samples,
                           float* __restrict__ out, int Nv)
{
    __shared__ __bf16 Ks [BN * KPAD];
    __shared__ __bf16 VTs[DD * 132];
    __shared__ float  Ss [BM * 132];
    __shared__ __bf16 Ps [BM * 132];
    __shared__ float  s2s[BN];
    __shared__ float  mrow[BM], lrow[BM], arow[BM], invden[BM], tsc[BM];

    const int tid  = threadIdx.x;
    const int wave = tid >> 5;
    const int lane = tid & 31;
    const int m0   = blockIdx.x * BM;
    const int mb    = wave & 1;
    const int dgrp  = wave >> 1;
    const int hi    = (lane >= 16) ? 1 : 0;
    const int l16   = lane & 15;
    const int rA    = mb * 16 + l16;
    const int koffA = hi * 8;

    if (tid < BM) {
        float tv = tarr[m0 + tid];
        float ts = 0.01f * __expf(tv * LN_RATIO);
        tsc[tid]    = ts;
        invden[tid] = 1.0f / (ts * ts + 1e-8f);
        mrow[tid]   = -__builtin_inff();
        lrow[tid]   = 0.0f;
    }
    if (tid < BN) s2s[tid] = 0.0f;

    v16bf Qa[8];
    {
        const float* qrow = x + (size_t)(m0 + rA) * DD;
#pragma unroll
        for (int kk = 0; kk < 8; ++kk) {
            const int k0 = kk * 32;
#pragma unroll
            for (int e = 0; e < 8; ++e) {
                Qa[kk][e]     = (__bf16)qrow[k0 + 2 * koffA + e];
                Qa[kk][e + 8] = (__bf16)qrow[k0 + 2 * koffA + 8 + e];
            }
        }
    }
    v8f Oacc[4];
    {
        v8f z = {};
#pragma unroll
        for (int dt = 0; dt < 4; ++dt) Oacc[dt] = z;
    }
    __syncthreads();
    const int ncol0 = dgrp * 32;

    for (int n0 = 0; n0 < Nv; n0 += BN) {
        for (int i = tid; i < BN * DD / 4; i += 256) {
            const int flat = i * 4;
            const int n = flat >> 8;
            const int c = flat & 255;
            const float4 f4 = *(const float4*)(samples + (size_t)(n0 + n) * DD + c);
            const __bf16 b0 = (__bf16)f4.x, b1 = (__bf16)f4.y;
            const __bf16 b2 = (__bf16)f4.z, b3 = (__bf16)f4.w;
            __bf16* kp = &Ks[n * KPAD + c];
            kp[0] = b0; kp[1] = b1; kp[2] = b2; kp[3] = b3;
            VTs[(c + 0) * 132 + n] = b0;
            VTs[(c + 1) * 132 + n] = b1;
            VTs[(c + 2) * 132 + n] = b2;
            VTs[(c + 3) * 132 + n] = b3;
            atomicAdd(&s2s[n], f4.x*f4.x + f4.y*f4.y + f4.z*f4.z + f4.w*f4.w);
        }
        __syncthreads();
#pragma unroll
        for (int tt = 0; tt < 2; ++tt) {
            const int ncol = ncol0 + tt * 16;
            v8f sacc = {};
            const __bf16* brow = &Ks[(ncol + l16) * KPAD + hi * 16];
#pragma unroll
            for (int kk = 0; kk < 8; ++kk) {
                v16bf b;
                const __bf16* bp = brow + kk * 32;
#pragma unroll
                for (int e = 0; e < 16; ++e) b[e] = bp[e];
                sacc = __builtin_amdgcn_wmma_f32_16x16x32_bf16(
                           false, Qa[kk], false, b, (short)0, sacc, false, false);
            }
#pragma unroll
            for (int vv = 0; vv < 8; ++vv)
                Ss[(mb * 16 + vv + hi * 8) * 132 + ncol + l16] = sacc[vv];
        }
        __syncthreads();
#pragma unroll
        for (int rr = 0; rr < 4; ++rr) {
            const int r = wave * 4 + rr;
            const float iv = invden[r];
            float v0 = (Ss[r * 132 + lane     ] - 0.5f * s2s[lane     ]) * iv;
            float v1 = (Ss[r * 132 + lane + 32] - 0.5f * s2s[lane + 32]) * iv;
            float v2 = (Ss[r * 132 + lane + 64] - 0.5f * s2s[lane + 64]) * iv;
            float v3 = (Ss[r * 132 + lane + 96] - 0.5f * s2s[lane + 96]) * iv;
            float tm = fmaxf(fmaxf(v0, v1), fmaxf(v2, v3));
#pragma unroll
            for (int off = 16; off >= 1; off >>= 1)
                tm = fmaxf(tm, __shfl_xor(tm, off, 32));
            const float mold = mrow[r];
            const float mnew = fmaxf(mold, tm);
            const float p0 = __expf(v0 - mnew), p1 = __expf(v1 - mnew);
            const float p2 = __expf(v2 - mnew), p3 = __expf(v3 - mnew);
            float ls = p0 + p1 + p2 + p3;
#pragma unroll
            for (int off = 16; off >= 1; off >>= 1)
                ls += __shfl_xor(ls, off, 32);
            if (lane == 0) {
                const float al = __expf(mold - mnew);
                lrow[r] = lrow[r] * al + ls;
                mrow[r] = mnew;
                arow[r] = al;
            }
            Ps[r * 132 + lane     ] = (__bf16)p0;
            Ps[r * 132 + lane + 32] = (__bf16)p1;
            Ps[r * 132 + lane + 64] = (__bf16)p2;
            Ps[r * 132 + lane + 96] = (__bf16)p3;
        }
        __syncthreads();
        if (tid < BN) s2s[tid] = 0.0f;
        float al[8];
#pragma unroll
        for (int vv = 0; vv < 8; ++vv) al[vv] = arow[mb * 16 + vv + hi * 8];
#pragma unroll
        for (int dt = 0; dt < 4; ++dt)
#pragma unroll
            for (int vv = 0; vv < 8; ++vv) Oacc[dt][vv] *= al[vv];
#pragma unroll
        for (int ks = 0; ks < 4; ++ks) {
            v16bf a;
            const __bf16* ap = &Ps[rA * 132 + ks * 32 + 2 * koffA];
#pragma unroll
            for (int e = 0; e < 8; ++e) { a[e] = ap[e]; a[e + 8] = ap[e + 8]; }
#pragma unroll
            for (int dt = 0; dt < 4; ++dt) {
                const int d = dgrp * 64 + dt * 16 + l16;
                const __bf16* bp = &VTs[d * 132 + ks * 32 + hi * 16];
                v16bf b;
#pragma unroll
                for (int e = 0; e < 16; ++e) b[e] = bp[e];
                Oacc[dt] = __builtin_amdgcn_wmma_f32_16x16x32_bf16(
                               false, a, false, b, (short)0, Oacc[dt], false, false);
            }
        }
        __syncthreads();
    }
#pragma unroll
    for (int dt = 0; dt < 4; ++dt) {
        const int d = dgrp * 64 + dt * 16 + l16;
#pragma unroll
        for (int vv = 0; vv < 8; ++vv) {
            const int r = mb * 16 + vv + hi * 8;
            const float mean = Oacc[dt][vv] / lrow[r];
            const float xv = x[(size_t)(m0 + r) * DD + d];
            out[(size_t)(m0 + r) * DD + d] = tsc[r] * (mean - xv) * invden[r];
        }
    }
}

extern "C" void kernel_launch(void* const* d_in, const int* in_sizes, int n_in,
                              void* d_out, int out_size, void* d_ws, size_t ws_size,
                              hipStream_t stream) {
    const float* t       = (const float*)d_in[0];
    const float* x       = (const float*)d_in[1];
    const float* samples = (const float*)d_in[2];
    float* out = (float*)d_out;

    const int B = in_sizes[0];          // 4096
    const int N = in_sizes[2] / DD;     // 8192

    const size_t kg_bytes  = (size_t)N * DD * sizeof(__bf16);   // 4 MB
    const size_t s2_bytes  = (size_t)N * sizeof(float);
    const size_t ws_needed = 2 * kg_bytes + s2_bytes;

    if (ws_size >= ws_needed) {
        __bf16* Kg  = (__bf16*)d_ws;
        __bf16* KTg = (__bf16*)((char*)d_ws + kg_bytes);
        float*  s2g = (float*)((char*)d_ws + 2 * kg_bytes);
        prepass_kernel<<<dim3(N / 64), dim3(256), 0, stream>>>(samples, Kg, KTg, s2g, N);
        score_kernel<<<dim3(B / BM), dim3(256), 0, stream>>>(t, x, Kg, KTg, s2g, out, N);
    } else {
        score_kernel_fallback<<<dim3(B / BM), dim3(256), 0, stream>>>(t, x, samples, out, N);
    }
}
